// VQ_26645977105136
// MI455X (gfx1250) — matrix operand order repaired and checked by reading, not compiled, not run on hardware
//
#include <hip/hip_runtime.h>

typedef _Float16 f16;
typedef __attribute__((ext_vector_type(16))) _Float16 v16h;
typedef __attribute__((ext_vector_type(8)))  _Float16 v8h;
typedef __attribute__((ext_vector_type(8)))  float    v8f;

union HF16 { v16h v; v8h h[2]; };

static __device__ __forceinline__ v16h frag_from(v8h lo, v8h hi) {
  HF16 u; u.h[0] = lo; u.h[1] = hi; return u.v;
}

static __device__ __forceinline__ v8f wmma_f16(v16h a, v16h b, v8f c) {
  return __builtin_amdgcn_wmma_f32_16x16x32_f16(false, a, false, b,
                                                (short)0, c, false, false);
}

// ---------------------------------------------------------------------------
// Elementwise / packing kernels
// ---------------------------------------------------------------------------
__global__ __launch_bounds__(256)
void k_cast_f16(const float* __restrict__ a, f16* __restrict__ o, size_t n) {
  size_t i = ((size_t)blockIdx.x * 256 + threadIdx.x) * 8;
  if (i + 8 <= n) {
    v8h r;
#pragma unroll
    for (int j = 0; j < 8; ++j) r[j] = (f16)a[i + j];
    *(v8h*)(o + i) = r;
  }
}

// torch conv weight (Cout, Cin, 4,4,4) -> f16 fragment-order [K/32][Cout][32]
__global__ void k_pack_conv_w(const float* __restrict__ w, f16* __restrict__ wq,
                              int Cout, int Cin) {
  int i = blockIdx.x * blockDim.x + threadIdx.x;
  int total = Cout * Cin * 64;
  if (i >= total) return;
  int t  = i & 63;
  int ci = (i >> 6) % Cin;
  int co = i / (Cin * 64);
  int k  = ci * 64 + t;
  wq[((size_t)(k >> 5) * Cout + co) * 32 + (k & 31)] = (f16)w[i];
}

// torch deconv weight (Cin, Cout, 4,4,4) -> f16 [8 parity][K/32][Cout][32]
__global__ void k_pack_deconv_w(const float* __restrict__ w, f16* __restrict__ wq,
                                int Cin, int Cout) {
  int i = blockIdx.x * blockDim.x + threadIdx.x;
  int total = Cin * Cout * 64;
  if (i >= total) return;
  int tap = i & 63;
  int co  = (i >> 6) % Cout;
  int ci  = i / (Cout * 64);
  int kd = tap >> 4, kh = (tap >> 2) & 3, kw = tap & 3;
  int pd = (kd + 1) & 1, td = kd >> 1;
  int ph = (kh + 1) & 1, th = kh >> 1;
  int pw = (kw + 1) & 1, tw = kw >> 1;
  int p = pd * 4 + ph * 2 + pw;
  int t = td * 4 + th * 2 + tw;
  int k = ci * 8 + t;
  int K = Cin * 8;
  wq[(size_t)p * K * Cout + ((size_t)(k >> 5) * Cout + co) * 32 + (k & 31)] =
      (f16)w[i];
}

__global__ void k_emb_pack(const float* __restrict__ emb, f16* __restrict__ embh,
                           float* __restrict__ embn) {
  int c = blockIdx.x * blockDim.x + threadIdx.x;
  if (c >= 512) return;
  float s = 0.f;
  for (int e = 0; e < 64; ++e) {
    float v = emb[c * 64 + e];
    s += v * v;
    embh[c * 64 + e] = (f16)v;
  }
  embn[c] = s;
}

// ---------------------------------------------------------------------------
// Implicit-GEMM conv3d, kernel 4, stride 2, pad 1, cubic dims.
// Block tile 128(M) x 64(N), 8 waves (4x2), each wave 32x32 = 4 WMMAs/K-step.
// K-loop unrolled by 64 (two 32-phases with constant tap tables).
// ---------------------------------------------------------------------------
__global__ __launch_bounds__(256)
void k_conv_s2(const f16* __restrict__ in, const f16* __restrict__ wq,
               const float* __restrict__ bias, f16* __restrict__ out,
               int Cin, int S, int Cout, int relu) {
  const int So = S >> 1;
  const int Mtot = So * So * So;
  const int K  = Cin * 64;
  const int S2 = S * S, S3 = S2 * S;
  __shared__ f16 As[2][128][40];

  const int tid = threadIdx.x;
  const int b   = blockIdx.z;
  const int m0  = blockIdx.x * 128;
  const int n0  = blockIdx.y * 64;

  const int arow = tid >> 2;        // 0..63; handles rows arow and arow+64
  const int aseg = (tid & 3) * 8;   // 0,8,16,24
  const f16* inB = in + (size_t)b * Cin * S3;

  // Per-thread tap tables: [rowpair][phase][j] (loop-invariant)
  unsigned Ts[2][2][8];
  f16      Ms[2][2][8];
#pragma unroll
  for (int rp = 0; rp < 2; ++rp) {
    int m  = m0 + arow + rp * 64;
    int od = m / (So * So);
    int r2 = m - od * (So * So);
    int oh = r2 / So;
    int ow = r2 - oh * So;
    int id0 = od * 2 - 1, ih0 = oh * 2 - 1, iw0 = ow * 2 - 1;
#pragma unroll
    for (int ph = 0; ph < 2; ++ph) {
#pragma unroll
      for (int j = 0; j < 8; ++j) {
        int rem = ph * 32 + aseg + j;
        int kd = rem >> 4, kh = (rem >> 2) & 3, kw = rem & 3;
        int id = id0 + kd, ih = ih0 + kh, iw = iw0 + kw;
        bool ok = ((unsigned)id < (unsigned)S) & ((unsigned)ih < (unsigned)S) &
                  ((unsigned)iw < (unsigned)S);
        Ts[rp][ph][j] = ok ? (unsigned)(id * S2 + ih * S + iw) : 0u;
        Ms[rp][ph][j] = ok ? (f16)1 : (f16)0;
      }
    }
  }

  const int wave = tid >> 5, lane = tid & 31;
  const int wm = wave >> 1, wn = wave & 1;
  const int lhalf = lane >> 4, l15 = lane & 15;

  const f16* wcol0 = wq + (size_t)(n0 + wn * 32 + l15) * 32 + (size_t)lhalf * 8;
  const f16* wcol1 = wcol0 + 16 * 32;

  v8f a00 = {0,0,0,0,0,0,0,0}, a01 = a00, a10 = a00, a11 = a00;
  int buf = 0;

  auto kstep = [&](int phase, int ciS3, int ktile) {
    v8h av0, av1;
#pragma unroll
    for (int j = 0; j < 8; ++j)
      av0[j] = inB[(unsigned)ciS3 + Ts[0][phase][j]] * Ms[0][phase][j];
#pragma unroll
    for (int j = 0; j < 8; ++j)
      av1[j] = inB[(unsigned)ciS3 + Ts[1][phase][j]] * Ms[1][phase][j];
    *(v8h*)&As[buf][arow][aseg]      = av0;
    *(v8h*)&As[buf][arow + 64][aseg] = av1;
    __syncthreads();

    const f16* a0p = &As[buf][wm * 32 + l15][lhalf * 8];
    const f16* a1p = &As[buf][wm * 32 + 16 + l15][lhalf * 8];
    v16h af0 = frag_from(*(const v8h*)a0p, *(const v8h*)(a0p + 16));
    v16h af1 = frag_from(*(const v8h*)a1p, *(const v8h*)(a1p + 16));
    const f16* bp0 = wcol0 + (size_t)ktile * Cout * 32;
    const f16* bp1 = wcol1 + (size_t)ktile * Cout * 32;
    v16h bf0 = frag_from(*(const v8h*)bp0, *(const v8h*)(bp0 + 16));
    v16h bf1 = frag_from(*(const v8h*)bp1, *(const v8h*)(bp1 + 16));
    __builtin_prefetch(bp0 + (size_t)Cout * 32, 0, 1);

    a00 = wmma_f16(af0, bf0, a00);
    a01 = wmma_f16(af0, bf1, a01);
    a10 = wmma_f16(af1, bf0, a10);
    a11 = wmma_f16(af1, bf1, a11);
    buf ^= 1;
  };

  for (int kb = 0; kb < K; kb += 64) {
    int ciS3 = (kb >> 6) * S3;
    kstep(0, ciS3, (kb >> 5));
    kstep(1, ciS3, (kb >> 5) + 1);
  }

  // epilogue
#pragma unroll
  for (int ni = 0; ni < 2; ++ni) {
    int n = n0 + wn * 32 + ni * 16 + l15;
    float bsv = bias[n];
    f16* oc = out + ((size_t)b * Cout + n) * Mtot;
#pragma unroll
    for (int mi = 0; mi < 2; ++mi) {
      v8f acc = (mi == 0) ? (ni == 0 ? a00 : a01) : (ni == 0 ? a10 : a11);
      int mbase = m0 + wm * 32 + mi * 16 + lhalf * 8;
#pragma unroll
      for (int v = 0; v < 8; ++v) {
        float r = acc[v] + bsv;
        if (relu) r = fmaxf(r, 0.f);
        oc[mbase + v] = (f16)r;
      }
    }
  }
}

// ---------------------------------------------------------------------------
// Implicit-GEMM ConvTranspose3d (k=4,s=2,p=1) via parity classes; K = Cin*8.
// Same 128x64 tile, 4 WMMAs per K-step, fixed tap tables (no phase).
// blockIdx.z = b*8 + parity.
// ---------------------------------------------------------------------------
__global__ __launch_bounds__(256)
void k_deconv_s2(const f16* __restrict__ in, const f16* __restrict__ wq,
                 const float* __restrict__ bias, f16* __restrict__ out,
                 int Cin, int S, int Cout, int relu) {
  const int So = S * 2;
  const int K  = Cin * 8;
  const int S2 = S * S, S3 = S2 * S;
  __shared__ f16 As[2][128][40];

  const int tid = threadIdx.x;
  const int bz  = blockIdx.z;
  const int p   = bz & 7, b = bz >> 3;
  const int pd = p >> 2, ph = (p >> 1) & 1, pw = p & 1;
  const int m0 = blockIdx.x * 128;
  const int n0 = blockIdx.y * 64;
  const f16* wpp = wq + (size_t)p * K * Cout;

  const int arow = tid >> 2;
  const int aseg = (tid & 3) * 8;
  const f16* inB = in + (size_t)b * Cin * S3;

  unsigned Ts[2][8];
  f16      Ms[2][8];
#pragma unroll
  for (int rp = 0; rp < 2; ++rp) {
    int m   = m0 + arow + rp * 64;
    int odp = m / S2;
    int r2  = m - odp * S2;
    int ohp = r2 / S;
    int owp = r2 - ohp * S;
#pragma unroll
    for (int j = 0; j < 8; ++j) {
      int td = (j >> 2) & 1, th = (j >> 1) & 1, tw = j & 1;
      int id = odp + pd - td, ih = ohp + ph - th, iw = owp + pw - tw;
      bool ok = ((unsigned)id < (unsigned)S) & ((unsigned)ih < (unsigned)S) &
                ((unsigned)iw < (unsigned)S);
      Ts[rp][j] = ok ? (unsigned)(id * S2 + ih * S + iw) : 0u;
      Ms[rp][j] = ok ? (f16)1 : (f16)0;
    }
  }

  const int wave = tid >> 5, lane = tid & 31;
  const int wm = wave >> 1, wn = wave & 1;
  const int lhalf = lane >> 4, l15 = lane & 15;

  const f16* wcol0 = wpp + (size_t)(n0 + wn * 32 + l15) * 32 + (size_t)lhalf * 8;
  const f16* wcol1 = wcol0 + 16 * 32;

  v8f a00 = {0,0,0,0,0,0,0,0}, a01 = a00, a10 = a00, a11 = a00;
  int buf = 0;

  for (int kb = 0; kb < K; kb += 32) {
    int ciS3 = ((kb + aseg) >> 3) * S3;
    v8h av0, av1;
#pragma unroll
    for (int j = 0; j < 8; ++j)
      av0[j] = inB[(unsigned)ciS3 + Ts[0][j]] * Ms[0][j];
#pragma unroll
    for (int j = 0; j < 8; ++j)
      av1[j] = inB[(unsigned)ciS3 + Ts[1][j]] * Ms[1][j];
    *(v8h*)&As[buf][arow][aseg]      = av0;
    *(v8h*)&As[buf][arow + 64][aseg] = av1;
    __syncthreads();

    const f16* a0p = &As[buf][wm * 32 + l15][lhalf * 8];
    const f16* a1p = &As[buf][wm * 32 + 16 + l15][lhalf * 8];
    v16h af0 = frag_from(*(const v8h*)a0p, *(const v8h*)(a0p + 16));
    v16h af1 = frag_from(*(const v8h*)a1p, *(const v8h*)(a1p + 16));
    const f16* bp0 = wcol0 + (size_t)(kb >> 5) * Cout * 32;
    const f16* bp1 = wcol1 + (size_t)(kb >> 5) * Cout * 32;
    v16h bf0 = frag_from(*(const v8h*)bp0, *(const v8h*)(bp0 + 16));
    v16h bf1 = frag_from(*(const v8h*)bp1, *(const v8h*)(bp1 + 16));
    __builtin_prefetch(bp0 + (size_t)Cout * 32, 0, 1);

    a00 = wmma_f16(af0, bf0, a00);
    a01 = wmma_f16(af0, bf1, a01);
    a10 = wmma_f16(af1, bf0, a10);
    a11 = wmma_f16(af1, bf1, a11);
    buf ^= 1;
  }

  const size_t So3 = (size_t)So * So * So;
#pragma unroll
  for (int ni = 0; ni < 2; ++ni) {
    int n = n0 + wn * 32 + ni * 16 + l15;
    float bsv = bias[n];
    f16* oc = out + ((size_t)b * Cout + n) * So3;
#pragma unroll
    for (int mi = 0; mi < 2; ++mi) {
      v8f acc = (mi == 0) ? (ni == 0 ? a00 : a01) : (ni == 0 ? a10 : a11);
      int mbase = m0 + wm * 32 + mi * 16 + lhalf * 8;
#pragma unroll
      for (int v = 0; v < 8; ++v) {
        int mm = mbase + v;
        int od_ = mm / S2;
        int rr  = mm - od_ * S2;
        int oh_ = rr / S;
        int ow_ = rr - oh_ * S;
        int od = od_ * 2 + pd, oh = oh_ * 2 + ph, ow = ow_ * 2 + pw;
        float r = acc[v] + bsv;
        if (relu) r = fmaxf(r, 0.f);
        oc[((size_t)od * So + oh) * So + ow] = (f16)r;
      }
    }
  }
}

// ---------------------------------------------------------------------------
// VQ nearest-code search via WMMA:  argmin_n (||e_n||^2 - 2 f.e_n)
// ---------------------------------------------------------------------------
__global__ __launch_bounds__(256)
void k_vq_argmin(const f16* __restrict__ h, const f16* __restrict__ embh,
                 const float* __restrict__ embn, int* __restrict__ idx) {
  const int tid  = threadIdx.x;
  const int wave = tid >> 5, lane = tid & 31;
  const int lhalf = lane >> 4, l15 = lane & 15;
  const int gw = blockIdx.x * 8 + wave;
  const int r0 = gw * 16;

  v16h a0, a1;
  {
    int r  = r0 + l15;
    int bb = r >> 14;
    int vx = (r >> 2) & 4095;
    int s  = r & 3;
    const f16* base = h + ((size_t)bb * 256 + s * 64) * 4096 + vx;
    int kbase = lhalf * 8;
#pragma unroll
    for (int j = 0; j < 8; ++j) {
      a0[j]     = base[(size_t)(kbase + j) * 4096];
      a0[8 + j] = base[(size_t)(16 + kbase + j) * 4096];
      a1[j]     = base[(size_t)(32 + kbase + j) * 4096];
      a1[8 + j] = base[(size_t)(48 + kbase + j) * 4096];
    }
  }

  float best[8];
  int   bidx[8];
#pragma unroll
  for (int v = 0; v < 8; ++v) { best[v] = 3.4e38f; bidx[v] = 0; }

  for (int nt = 0; nt < 32; ++nt) {
    const int code = nt * 16 + l15;
    const v8h* e0 = (const v8h*)(embh + code * 64 + lhalf * 8);
    const v8h* e1 = (const v8h*)(embh + code * 64 + 32 + lhalf * 8);
    v16h b0 = frag_from(e0[0], e0[2]);
    v16h b1 = frag_from(e1[0], e1[2]);

    v8f acc = {0.f, 0.f, 0.f, 0.f, 0.f, 0.f, 0.f, 0.f};
    acc = wmma_f16(a0, b0, acc);
    acc = wmma_f16(a1, b1, acc);
    const float en = embn[code];
#pragma unroll
    for (int v = 0; v < 8; ++v) {
      float d = en - 2.f * acc[v];
      if (d < best[v]) { best[v] = d; bidx[v] = code; }
    }
  }

#pragma unroll
  for (int v = 0; v < 8; ++v) {
    float bv = best[v];
    int   bi = bidx[v];
    for (int off = 8; off >= 1; off >>= 1) {
      float ov = __shfl_xor(bv, off, 32);
      int   oi = __shfl_xor(bi, off, 32);
      if (ov < bv) { bv = ov; bi = oi; }
    }
    if (l15 == 0) idx[r0 + lhalf * 8 + v] = bi;
  }
}

// ---------------------------------------------------------------------------
// Gather quantized codes into NCDHW f16 + accumulate 1.25*mean((q-h)^2)
// ---------------------------------------------------------------------------
__global__ __launch_bounds__(256)
void k_quant_loss(const f16* __restrict__ h, const f16* __restrict__ embh,
                  const int* __restrict__ idx, f16* __restrict__ q,
                  float* __restrict__ loss) {
  __shared__ float red[256];
  size_t i = (size_t)blockIdx.x * 256 + threadIdx.x;
  int vx = (int)(i & 4095);
  int c  = (int)((i >> 12) & 255);
  int b  = (int)(i >> 20);
  int r  = (b * 4096 + vx) * 4 + (c >> 6);
  int e  = c & 63;
  float qv = (float)embh[idx[r] * 64 + e];
  float hv = (float)h[i];
  q[i] = (f16)qv;
  float d = qv - hv;
  red[threadIdx.x] = d * d;
  __syncthreads();
  for (int s = 128; s > 0; s >>= 1) {
    if (threadIdx.x < (unsigned)s) red[threadIdx.x] += red[threadIdx.x + s];
    __syncthreads();
  }
  if (threadIdx.x == 0) atomicAdd(loss, red[0] * (1.25f / 2097152.f));
}

// ---------------------------------------------------------------------------
// Final 1x1x1 conv: 128 channels -> 1, f32 out; 8 voxels per thread
// ---------------------------------------------------------------------------
__global__ __launch_bounds__(256)
void k_conv1x1_out(const f16* __restrict__ in, const float* __restrict__ wo,
                   const float* __restrict__ bo, float* __restrict__ out) {
  const int S3 = 2097152;  // 128^3
  int t = blockIdx.x * 256 + threadIdx.x;
  int b = t >> 18;
  int p8 = (t & 262143) * 8;
  const f16* src = in + (size_t)b * 128 * S3 + p8;
  float s[8];
  float bv = bo[0];
#pragma unroll
  for (int j = 0; j < 8; ++j) s[j] = bv;
  for (int c = 0; c < 128; ++c) {
    v8h d = *(const v8h*)(src + (size_t)c * S3);
    float w = wo[c];
#pragma unroll
    for (int j = 0; j < 8; ++j) s[j] += (float)d[j] * w;
  }
  float* dst = out + (size_t)b * S3 + p8;
#pragma unroll
  for (int j = 0; j < 8; ++j) dst[j] = s[j];
}

// ---------------------------------------------------------------------------
extern "C" void kernel_launch(void* const* d_in, const int* in_sizes, int n_in,
                              void* d_out, int out_size, void* d_ws, size_t ws_size,
                              hipStream_t stream) {
  (void)in_sizes; (void)n_in; (void)out_size; (void)ws_size;
  const float* x   = (const float*)d_in[0];
  const float* w1  = (const float*)d_in[1];
  const float* b1  = (const float*)d_in[2];
  const float* w2  = (const float*)d_in[3];
  const float* b2  = (const float*)d_in[4];
  const float* w3  = (const float*)d_in[5];
  const float* b3  = (const float*)d_in[6];
  const float* dw1 = (const float*)d_in[7];
  const float* db1 = (const float*)d_in[8];
  const float* dw2 = (const float*)d_in[9];
  const float* db2 = (const float*)d_in[10];
  const float* dw3 = (const float*)d_in[11];
  const float* db3 = (const float*)d_in[12];
  const float* wo  = (const float*)d_in[13];
  const float* bo  = (const float*)d_in[14];
  const float* emb = (const float*)d_in[15];

  char* ws = (char*)d_ws;
  size_t off = 0;
  auto carve = [&](size_t bytes) -> void* {
    void* pp = ws + off;
    off = (off + bytes + 255) & ~(size_t)255;
    return pp;
  };
  f16* xh   = (f16*)carve(4194304ull * 2);     // [2,1,128^3]
  f16* a1   = (f16*)carve(67108864ull * 2);    // [2,128,64^3]
  f16* a2   = (f16*)carve(16777216ull * 2);    // [2,256,32^3]
  f16* h16  = (f16*)carve(2097152ull * 2);     // [2,256,16^3]
  f16* q16  = (f16*)carve(2097152ull * 2);
  f16* dd1  = (f16*)carve(16777216ull * 2);    // [2,256,32^3]
  f16* dd2  = (f16*)carve(67108864ull * 2);    // [2,128,64^3]
  f16* dd3  = (f16*)carve(536870912ull * 2);   // [2,128,128^3]
  f16* w1p  = (f16*)carve(8192ull * 2);
  f16* w2p  = (f16*)carve(2097152ull * 2);
  f16* w3p  = (f16*)carve(4194304ull * 2);
  f16* dw1p = (f16*)carve(4194304ull * 2);
  f16* dw2p = (f16*)carve(2097152ull * 2);
  f16* dw3p = (f16*)carve(1048576ull * 2);
  f16* embh = (f16*)carve(32768ull * 2);
  float* embn = (float*)carve(512ull * 4);
  int* idxb   = (int*)carve(32768ull * 4);

  hipMemsetAsync(d_out, 0, sizeof(float), stream);  // loss accumulator

  // packing / casts
  k_cast_f16<<<2048, 256, 0, stream>>>(x, xh, 4194304ull);
  k_pack_conv_w<<<(128 * 1 * 64 + 255) / 256, 256, 0, stream>>>(w1, w1p, 128, 1);
  k_pack_conv_w<<<(256 * 128 * 64 + 255) / 256, 256, 0, stream>>>(w2, w2p, 256, 128);
  k_pack_conv_w<<<(256 * 256 * 64 + 255) / 256, 256, 0, stream>>>(w3, w3p, 256, 256);
  k_pack_deconv_w<<<(256 * 256 * 64 + 255) / 256, 256, 0, stream>>>(dw1, dw1p, 256, 256);
  k_pack_deconv_w<<<(256 * 128 * 64 + 255) / 256, 256, 0, stream>>>(dw2, dw2p, 256, 128);
  k_pack_deconv_w<<<(128 * 128 * 64 + 255) / 256, 256, 0, stream>>>(dw3, dw3p, 128, 128);
  k_emb_pack<<<2, 256, 0, stream>>>(emb, embh, embn);

  // encoder (tiles: 128 M x 64 N)
  k_conv_s2<<<dim3(2048, 2, 2), 256, 0, stream>>>(xh, w1p, b1, a1, 1, 128, 128, 1);
  k_conv_s2<<<dim3(256, 4, 2), 256, 0, stream>>>(a1, w2p, b2, a2, 128, 64, 256, 1);
  k_conv_s2<<<dim3(32, 4, 2), 256, 0, stream>>>(a2, w3p, b3, h16, 256, 32, 256, 0);

  // vector quantization + loss
  k_vq_argmin<<<256, 256, 0, stream>>>(h16, embh, embn, idxb);
  k_quant_loss<<<8192, 256, 0, stream>>>(h16, embh, idxb, q16, (float*)d_out);

  // decoder (blockIdx.z = b*8 + parity)
  k_deconv_s2<<<dim3(32, 4, 16), 256, 0, stream>>>(q16, dw1p, db1, dd1, 256, 16, 256, 1);
  k_deconv_s2<<<dim3(256, 2, 16), 256, 0, stream>>>(dd1, dw2p, db2, dd2, 256, 32, 128, 1);
  k_deconv_s2<<<dim3(2048, 2, 16), 256, 0, stream>>>(dd2, dw3p, db3, dd3, 128, 64, 128, 0);

  // output head
  k_conv1x1_out<<<2048, 256, 0, stream>>>(dd3, wo, bo, ((float*)d_out) + 1);
}
